// SVDQLinear_77910706749961
// MI455X (gfx1250) — matrix operand order repaired and checked by reading, not compiled, not run on hardware
//
#include <hip/hip_runtime.h>
#include <hip/hip_bf16.h>
#include <hip/hip_fp16.h>
#include <cstdint>

typedef __attribute__((ext_vector_type(16))) __bf16 v16bf;
typedef __attribute__((ext_vector_type(8)))  float  v8f;

union FragAB {
    v16bf v;
    uint4 q[2];
    uint32_t u[8];
};

__device__ __forceinline__ uint32_t pack2_bf16(float lo, float hi) {
    union { __bf16 b[2]; uint32_t u; } r;
    r.b[0] = (__bf16)lo;
    r.b[1] = (__bf16)hi;
    return r.u;
}

// ---- NVFP4 (e2m1) -> packed bf16 via v_perm_b32 byte-table lookups ----
// bf16 magnitude patterns for codes 0..7 (0,0.5,1,1.5,2,3,4,6):
//   lo bytes L[8] = 00 00 80 C0 00 40 80 C0
//   hi bytes H[8] = 00 3F 3F 3F 40 40 40 40
// Dequants one dword (8 codes) into 4 output dwords (2 bf16 each), k-ascending,
// scaled by sc in f32 (lowers to v_fma_mixlo/hi_bf16).
__device__ __forceinline__ void dequant8(uint32_t w, float sc, uint32_t* out) {
    const uint32_t Ltab_lo = 0xC0800000u, Ltab_hi = 0xC0804000u;  // L[0..3], L[4..7]
    const uint32_t Htab_lo = 0x3F3F3F00u, Htab_hi = 0x40404040u;  // H[0..3], H[4..7]
    uint32_t mlo = w & 0x07070707u;          // m3 of even codes, byte-spread
    uint32_t mhi = (w >> 4) & 0x07070707u;   // m3 of odd codes
    uint32_t slo = (w << 4) & 0x80808080u;   // sign of even codes -> bit7 per byte
    uint32_t shi = w & 0x80808080u;          // sign of odd codes  -> bit7 per byte
#pragma unroll
    for (uint32_t j = 0; j < 4; ++j) {
        uint32_t CL = 0x0C040C00u + j * 0x00010001u;   // [j, 0x00, 4+j, 0x00]
        uint32_t CH = 0x040C000Cu + j * 0x01000100u;   // [0x00, j, 0x00, 4+j]
        uint32_t selL = __builtin_amdgcn_perm(mhi, mlo, CL);  // [m3lo,0,m3hi,0]
        uint32_t selH = __builtin_amdgcn_perm(mhi, mlo, CH);  // [0,m3lo,0,m3hi]
        uint32_t lv = __builtin_amdgcn_perm(Ltab_hi, Ltab_lo, selL);
        uint32_t hv = __builtin_amdgcn_perm(Htab_hi, Htab_lo, selH);
        uint32_t sm = __builtin_amdgcn_perm(shi, slo, CH);    // signs -> bits 15,31
        uint32_t bits = lv | hv | sm;        // packed exact bf16 levels
        float fl = __uint_as_float(bits << 16);          // even-k level (exact f32)
        float fh = __uint_as_float(bits & 0xFFFF0000u);  // odd-k level  (exact f32)
        out[j] = pack2_bf16(fl * sc, fh * sc);
    }
}

// Dequant 8 bytes (16 fp4 codes, one scale group) into a K-striped B fragment.
__device__ __forceinline__ void dequant16(uint2 wb, float sc, FragAB& f) {
    dequant8(wb.x, sc, &f.u[0]);
    dequant8(wb.y, sc, &f.u[4]);
}

// ---------------------------------------------------------------- kernel 1
__global__ __launch_bounds__(256) void prep_kernel(
    const __hip_bfloat16* __restrict__ x, const float* __restrict__ smooth,
    __hip_bfloat16* __restrict__ xs, int IN, int total)
{
    int i = blockIdx.x * blockDim.x + threadIdx.x;
    if (i >= total) return;
    int col = i & (IN - 1);   // IN is a power of two (8192)
    xs[i] = __float2bfloat16(__bfloat162float(x[i]) * smooth[col]);
}

// ---------------------------------------------------------------- kernel 2
// zt[b][r] for r in [0, RANK+LR): rows of core1 then lora_A. One wave per dot.
__global__ __launch_bounds__(256) void zt_kernel(
    const __hip_bfloat16* __restrict__ xs, const __hip_bfloat16* __restrict__ core1,
    const __hip_bfloat16* __restrict__ loraA, float* __restrict__ zt,
    int IN, int Bb, int RANK, int R2)
{
    int widx = (int)((blockIdx.x * blockDim.x + threadIdx.x) >> 5);
    int lane = threadIdx.x & 31;
    if (widx >= Bb * R2) return;
    int b = widx / R2, r = widx % R2;
    const __hip_bfloat16* wrow = (r < RANK) ? (core1 + (size_t)r * IN)
                                            : (loraA + (size_t)(r - RANK) * IN);
    const __hip_bfloat16* xrow = xs + (size_t)b * IN;
    float acc = 0.f;
    for (int k = lane; k < IN; k += 32)
        acc += __bfloat162float(xrow[k]) * __bfloat162float(wrow[k]);
    for (int m = 16; m; m >>= 1) acc += __shfl_xor(acc, m, 32);
    if (lane == 0)  // match reference's bf16 intermediate rounding of z / tmp
        zt[widx] = __bfloat162float(__float2bfloat16(acc));
}

// ---------------------------------------------------------------- kernel 3
// y0[b][o] = z.core0[o,:] + (alpha/LR)*tmp.lora_B[o,:] + bias[o]   (f32)
__global__ __launch_bounds__(256) void y0_kernel(
    const float* __restrict__ zt, const __hip_bfloat16* __restrict__ core0,
    const __hip_bfloat16* __restrict__ loraB, const __hip_bfloat16* __restrict__ bias,
    const float* __restrict__ lalpha, float* __restrict__ y0,
    int OUT, int Bb, int RANK, int LR)
{
    int idx = blockIdx.x * blockDim.x + threadIdx.x;
    if (idx >= OUT * Bb) return;
    int o = idx / Bb, b = idx % Bb;
    const float* z = zt + (size_t)b * (RANK + LR);
    float acc = 0.f;
    for (int r = 0; r < RANK; ++r)
        acc += z[r] * __bfloat162float(core0[(size_t)o * RANK + r]);
    float acc2 = 0.f;
    for (int j = 0; j < LR; ++j)
        acc2 += z[RANK + j] * __bfloat162float(loraB[(size_t)o * LR + j]);
    float ls = lalpha[0] / (float)(LR > 0 ? LR : 1);
    y0[(size_t)b * OUT + o] = acc + ls * acc2 + __bfloat162float(bias[o]);
}

// ---------------------------------------------------------------- kernel 4
// Main NVFP4 dequant GEMM with WMMA. Block = 128 threads = 4 waves; each wave
// owns a 16-wide N tile and the full K reduction, 64 K per trip (4 WMMAs).
// IN_T/OUT_T are compile-time strides (0 = use runtime args).
template<int IN_T, int OUT_T>
__global__ __launch_bounds__(128) void fp4_wmma_kernel(
    const __hip_bfloat16* __restrict__ xs, const unsigned char* __restrict__ w4,
    const __half* __restrict__ wscale, const float* __restrict__ wscale2,
    const float* __restrict__ y0, __hip_bfloat16* __restrict__ out,
    int in_rt, int out_rt)
{
    const int IN  = IN_T  ? IN_T  : in_rt;
    const int OUT = OUT_T ? OUT_T : out_rt;

    const int lane  = threadIdx.x & 31;
    const int wave  = threadIdx.x >> 5;
    const int nbase = blockIdx.x * 64 + wave * 16;
    if (nbase + 16 > OUT) return;            // uniform per wave (EXEC stays all-1)
    const int l15 = lane & 15;
    const int hi  = lane >> 4;               // half-wave selector (0/1)
    const int orow = nbase + l15;            // W row (output column) this lane feeds

    const float s2 = wscale2[0];
    const unsigned char* wrow = w4 + (size_t)orow * (size_t)(IN >> 1) + (size_t)(hi * 8);
    const __half* srow = wscale + (size_t)orow * (size_t)(IN >> 4) + hi;
    const __hip_bfloat16* xr0 = xs + (size_t)l15 * IN        + hi * 16;
    const __hip_bfloat16* xr1 = xs + (size_t)(l15 + 16) * IN + hi * 16;

    v8f c0 = {}; v8f c1 = {};
    const int nK2 = IN >> 6;                 // super-tiles of 64 K
    for (int kt2 = 0; kt2 < nK2; ++kt2) {
        const int kb = kt2 << 5;             // byte offset of super-tile in W stream
        const int ke = kt2 << 6;             // element offset of super-tile in xs
        const int kg = kt2 << 2;             // scale-group offset (4 groups / 64 K)

        // ---- weight + scale loads for both 32-K tiles (issued up front) ----
        uint2 wbA = *(const uint2*)(wrow + kb);
        uint2 wbB = *(const uint2*)(wrow + kb + 16);
        __builtin_prefetch(wrow + kb + 256, 0, 1);        // global_prefetch_b8
        float scA = __half2float(srow[kg])     * s2;
        float scB = __half2float(srow[kg + 2]) * s2;

        // ---- A fragments (K-contiguous layout, k = 16*hi + s), 8x b128 ----
        FragAB a0, a1, a2, a3;
        a0.q[0] = *(const uint4*)(xr0 + ke);
        a0.q[1] = *(const uint4*)(xr0 + ke + 8);
        a1.q[0] = *(const uint4*)(xr1 + ke);
        a1.q[1] = *(const uint4*)(xr1 + ke + 8);
        a2.q[0] = *(const uint4*)(xr0 + ke + 32);
        a2.q[1] = *(const uint4*)(xr0 + ke + 40);
        a3.q[0] = *(const uint4*)(xr1 + ke + 32);
        a3.q[1] = *(const uint4*)(xr1 + ke + 40);

        // ---- tile 0: K = ke..ke+31 ----
        FragAB bf0;
        dequant16(wbA, scA, bf0);
        c0 = __builtin_amdgcn_wmma_f32_16x16x32_bf16(false, a0.v, false, bf0.v,
                                                     (short)0, c0, false, false);
        c1 = __builtin_amdgcn_wmma_f32_16x16x32_bf16(false, a1.v, false, bf0.v,
                                                     (short)0, c1, false, false);

        // ---- tile 1: K = ke+32..ke+63 ----
        FragAB bf1;
        dequant16(wbB, scB, bf1);
        c0 = __builtin_amdgcn_wmma_f32_16x16x32_bf16(false, a2.v, false, bf1.v,
                                                     (short)0, c0, false, false);
        c1 = __builtin_amdgcn_wmma_f32_16x16x32_bf16(false, a3.v, false, bf1.v,
                                                     (short)0, c1, false, false);
    }

    // ---- epilogue: add low-rank/LoRA/bias partial and store bf16 ----
    const int n = nbase + l15;
    const float* y0p = y0 + (size_t)(hi * 8) * OUT + n;         // rows m = hi*8 + i
    __hip_bfloat16* outp = out + (size_t)(hi * 8) * OUT + n;
#pragma unroll
    for (int i = 0; i < 8; ++i) {
        size_t r0 = (size_t)i * OUT;                            // frag0: m = hi*8+i
        size_t r1 = (size_t)(i + 16) * OUT;                     // frag1: m + 16
        outp[r0] = __float2bfloat16(c0[i] + y0p[r0]);
        outp[r1] = __float2bfloat16(c1[i] + y0p[r1]);
    }
}

// ---------------------------------------------------------------- launcher
extern "C" void kernel_launch(void* const* d_in, const int* in_sizes, int n_in,
                              void* d_out, int out_size, void* d_ws, size_t ws_size,
                              hipStream_t stream) {
    const __hip_bfloat16* x      = (const __hip_bfloat16*)d_in[0];
    const __hip_bfloat16* core0  = (const __hip_bfloat16*)d_in[1];
    const __hip_bfloat16* core1  = (const __hip_bfloat16*)d_in[2];
    const unsigned char*  w4     = (const unsigned char*) d_in[3];
    const __half*         wscale = (const __half*)        d_in[4];
    const float*          wsc2   = (const float*)         d_in[5];
    const float*          smooth = (const float*)         d_in[6];
    const __hip_bfloat16* loraA  = (const __hip_bfloat16*)d_in[7];
    const __hip_bfloat16* loraB  = (const __hip_bfloat16*)d_in[8];
    const float*          lalpha = (const float*)         d_in[9];
    const __hip_bfloat16* bias   = (const __hip_bfloat16*)d_in[10];
    __hip_bfloat16* out = (__hip_bfloat16*)d_out;

    const int IN   = in_sizes[6];                 // 8192
    const int OUT  = in_sizes[10];                // 8192
    const int Bb   = in_sizes[0] / IN;            // 32
    const int RANK = in_sizes[1] / OUT;           // 64
    const int LR   = in_sizes[8] / OUT;           // 16
    const int R2   = RANK + LR;                   // 80

    char* base = (char*)d_ws;
    __hip_bfloat16* xs = (__hip_bfloat16*)base;                     // Bb*IN bf16
    size_t off = ((size_t)Bb * IN * 2 + 255) & ~(size_t)255;
    float* zt = (float*)(base + off);                                // Bb*R2 f32
    off = (off + (size_t)Bb * R2 * 4 + 255) & ~(size_t)255;
    float* y0 = (float*)(base + off);                                // Bb*OUT f32

    {   // xs = x * smooth
        int total = Bb * IN;
        prep_kernel<<<(total + 255) / 256, 256, 0, stream>>>(x, smooth, xs, IN, total);
    }
    {   // zt = xs @ [core1; lora_A]^T
        int waves = Bb * R2;
        zt_kernel<<<(waves + 7) / 8, 256, 0, stream>>>(xs, core1, loraA, zt, IN, Bb, RANK, R2);
    }
    {   // y0 = low-rank + lora + bias
        int total = OUT * Bb;
        y0_kernel<<<(total + 255) / 256, 256, 0, stream>>>(zt, core0, loraB, bias, lalpha,
                                                           y0, OUT, Bb, RANK, LR);
    }
    {   // y = xs @ dequant(W)^T + y0   (WMMA bf16)
        dim3 grid(OUT / 64), block(128);
        if (IN == 8192 && OUT == 8192) {
            fp4_wmma_kernel<8192, 8192><<<grid, block, 0, stream>>>(
                xs, w4, wscale, wsc2, y0, out, IN, OUT);
        } else {
            fp4_wmma_kernel<0, 0><<<grid, block, 0, stream>>>(
                xs, w4, wscale, wsc2, y0, out, IN, OUT);
        }
    }
}
